// RDGCN_68573447848483
// MI455X (gfx1250) — matrix operand order repaired and checked by reading, not compiled, not run on hardware
//
#include <hip/hip_runtime.h>
#include <math.h>

// Problem constants (from reference)
#define E_ENT  50000
#define R_REL  500
#define N_EDGE 500000
#define DIMD   256
#define DIM2   512
#define HIDN   128
#define ALPHA_C 0.1f
#define BETA_C  0.3f
#define SLOPE_C 0.01f

typedef __attribute__((ext_vector_type(16))) __bf16 v16bf;
typedef __attribute__((ext_vector_type(8)))  float  v8f;
typedef __attribute__((ext_vector_type(4)))  float  f32x4;
typedef __attribute__((ext_vector_type(4)))  int    v4i;

static __device__ __forceinline__ float leakyf(float x) {
  return x > 0.f ? x : SLOPE_C * x;
}

static __device__ __forceinline__ float waveSum(float v) {
#pragma unroll
  for (int off = 16; off > 0; off >>= 1) v += __shfl_down(v, off, 32);
  return v;
}

// float atomic max via int/uint ordering trick (handles mixed signs; init -inf)
static __device__ __forceinline__ float atomicMaxF(float* addr, float val) {
  if (val >= 0.f)
    return __int_as_float(atomicMax((int*)addr, __float_as_int(val)));
  return __uint_as_float(atomicMin((unsigned int*)addr, __float_as_uint(val)));
}

// ---------------------------------------------------------------------------
// Async global->LDS staging (gfx1250 GLOBAL_LOAD_ASYNC_TO_LDS_B128, ASYNCcnt).
// Probe-discovered signature: (int4 AS(1)*, int4 AS(3)*, imm offset, imm cpol).
// ---------------------------------------------------------------------------
static __device__ __forceinline__ void stage_b128(const float* g, float* l) {
#if __has_builtin(__builtin_amdgcn_global_load_async_to_lds_b128)
  __builtin_amdgcn_global_load_async_to_lds_b128(
      (__attribute__((address_space(1))) v4i*)g,
      (__attribute__((address_space(3))) v4i*)l, 0, 0);
#else
  *(f32x4*)l = *(const f32x4*)g;
#endif
}

static __device__ __forceinline__ void stage_wait() {
#if __has_builtin(__builtin_amdgcn_global_load_async_to_lds_b128)
#if __has_builtin(__builtin_amdgcn_s_wait_asynccnt)
  __builtin_amdgcn_s_wait_asynccnt(0);
#else
  asm volatile("s_wait_asynccnt 0x0" ::: "memory");
#endif
#endif
}

// ---------------------------------------------------------------------------
// fill
// ---------------------------------------------------------------------------
__global__ void fill_f32(float* __restrict__ p, long n, float v) {
  long i = (long)blockIdx.x * blockDim.x + threadIdx.x;
  if (i < n) p[i] = v;
}

// ---------------------------------------------------------------------------
// column sums of head_r / tail_r  (shared across both _compute_r calls)
// ---------------------------------------------------------------------------
__global__ void colsum2(const float* __restrict__ H, const float* __restrict__ T,
                        float* __restrict__ hs, float* __restrict__ ts) {
  int c = blockIdx.x * blockDim.x + threadIdx.x;
  if (c >= R_REL) return;
  int chunk = (E_ENT + gridDim.y - 1) / gridDim.y;
  int r0 = blockIdx.y * chunk;
  int r1 = min(r0 + chunk, E_ENT);
  float sh = 0.f, st = 0.f;
  for (int e = r0; e < r1; ++e) {
    sh += H[(long)e * R_REL + c];
    st += T[(long)e * R_REL + c];
  }
  atomicAdd(&hs[c], sh);
  atomicAdd(&ts[c], st);
}

// ---------------------------------------------------------------------------
// C[m, colOff+n] += sum_k A[k,m] * B[k,n]   A:[E_ENT x R_REL], B:[E_ENT x 256]
// Compile-time strides -> pointer-increment + constant instruction offsets.
// Two N-tiles per wave share one A fragment (A reuse, 2x WMMA ILP).
// K split across blockIdx.z; f32 atomic epilogue into pre-zeroed C[.,512].
// ---------------------------------------------------------------------------
__global__ void __launch_bounds__(32)
wmma_atb_atomic(const float* __restrict__ A, const float* __restrict__ B,
                float* __restrict__ C, int colOff, int kPerBlk) {
  constexpr int M = R_REL, K = E_ENT, lda = R_REL, ldb = DIMD, ldc = DIM2;
  const int lane = threadIdx.x;
  const bool hi = lane >= 16;
  const int l15 = lane & 15;
  const int m0 = blockIdx.x * 16;
  const int n0 = blockIdx.y * 32;                // two 16-wide tiles
  const int kbeg = blockIdx.z * kPerBlk;
  const int kend = min(kbeg + kPerBlk, K);
  const int mA = min(m0 + l15, M - 1);           // branch-free row clamp
  // bases advance by 32*ld per iteration; element offsets are constants
  const float* Ap = A + mA + (long)(kbeg + (hi ? 16 : 0)) * lda;
  const float* Bp0 = B + n0 + l15 + (long)(kbeg + (hi ? 16 : 0)) * ldb;
  const float* Bp1 = Bp0 + 16;
  v8f acc0 = {}, acc1 = {};
  int k0 = kbeg;
#pragma unroll 2
  for (; k0 + 32 <= kend; k0 += 32) {
    v16bf a, b0, b1;
#pragma unroll
    for (int j = 0; j < 8; ++j) {
      a[j]     = (__bf16)Ap[j * lda];            // ka0 + j
      a[j + 8] = (__bf16)Ap[(8 + j) * lda];      // ka0 + 8 + j
    }
#pragma unroll
    for (int j = 0; j < 16; ++j) {
      b0[j] = (__bf16)Bp0[j * ldb];
      b1[j] = (__bf16)Bp1[j * ldb];
    }
    acc0 = __builtin_amdgcn_wmma_f32_16x16x32_bf16(false, a, false, b0,
                                                   (short)0, acc0, false, false);
    acc1 = __builtin_amdgcn_wmma_f32_16x16x32_bf16(false, a, false, b1,
                                                   (short)0, acc1, false, false);
    Ap  += 32 * lda;
    Bp0 += 32 * ldb;
    Bp1 += 32 * ldb;
  }
  if (k0 < kend) {                               // guarded tail (last z-slice only)
    v16bf a, b0, b1;
#pragma unroll
    for (int j = 0; j < 16; ++j) {
      int ka = k0 + (hi ? 16 : 0) + j;
      int aoff = j * lda;
      a[j] = (__bf16)((ka < kend) ? Ap[aoff] : 0.f);
      int kb = k0 + (hi ? 16 : 0) + j;
      b0[j] = (__bf16)((kb < kend) ? Bp0[j * ldb] : 0.f);
      b1[j] = (__bf16)((kb < kend) ? Bp1[j * ldb] : 0.f);
    }
    acc0 = __builtin_amdgcn_wmma_f32_16x16x32_bf16(false, a, false, b0,
                                                   (short)0, acc0, false, false);
    acc1 = __builtin_amdgcn_wmma_f32_16x16x32_bf16(false, a, false, b1,
                                                   (short)0, acc1, false, false);
  }
#pragma unroll
  for (int j = 0; j < 8; ++j) {
    int m = m0 + j + (hi ? 8 : 0);
    if (m < M) {
      atomicAdd(&C[(long)m * ldc + colOff + n0 + l15], acc0[j]);
      atomicAdd(&C[(long)m * ldc + colOff + n0 + 16 + l15], acc1[j]);
    }
  }
}

// scale compute_r output rows: cols 0..255 /= hs[r], cols 256..511 /= ts[r]
__global__ void scale_dx(float* __restrict__ dX, const float* __restrict__ hs,
                         const float* __restrict__ ts) {
  int r = blockIdx.x;
  int c = threadIdx.x;
  float s = (c < DIMD) ? hs[r] : ts[r];
  dX[(long)r * DIM2 + c] /= s;
}

// ---------------------------------------------------------------------------
// C = relu(A @ B)   A: [M,512] row-major (b128 loads), B: [512,512].
// B column slice (512x16) staged to LDS via async b128; ds fragment reads.
// ---------------------------------------------------------------------------
__global__ void __launch_bounds__(32)
wmma_gemm_relu(const float* __restrict__ A, const float* __restrict__ B,
               float* __restrict__ C, int M) {
  __shared__ float Bs[DIM2 * 16];                // [k][n] slice, 32KB
  const int lane = threadIdx.x;
  const bool hi = lane >= 16;
  const int l15 = lane & 15;
  const int m0 = blockIdx.x * 16;
  const int n0 = blockIdx.y * 16;
  // stage B[:, n0:n0+16] -> Bs  (2048 float4s, 64 per lane)
  for (int i = lane; i < DIM2 * 4; i += 32) {
    int row = i >> 2, part = (i & 3) * 4;
    stage_b128(B + (long)row * DIM2 + n0 + part, Bs + row * 16 + part);
  }
  stage_wait();
  const int mA = min(m0 + l15, M - 1);
  const float* Arow = A + (long)mA * DIM2;
  v8f acc = {};
#pragma unroll 2
  for (int k0 = 0; k0 < DIM2; k0 += 32) {
    const int kaLo = k0 + (hi ? 16 : 0);
    const int kaHi = k0 + (hi ? 24 : 8);
    f32x4 a0 = *(const f32x4*)(Arow + kaLo);
    f32x4 a1 = *(const f32x4*)(Arow + kaLo + 4);
    f32x4 a2 = *(const f32x4*)(Arow + kaHi);
    f32x4 a3 = *(const f32x4*)(Arow + kaHi + 4);
    v16bf a, b;
#pragma unroll
    for (int j = 0; j < 4; ++j) {
      a[j]      = (__bf16)a0[j];
      a[j + 4]  = (__bf16)a1[j];
      a[j + 8]  = (__bf16)a2[j];
      a[j + 12] = (__bf16)a3[j];
    }
    const int kb0 = k0 + (hi ? 16 : 0);
#pragma unroll
    for (int j = 0; j < 16; ++j)
      b[j] = (__bf16)Bs[(kb0 + j) * 16 + l15];
    acc = __builtin_amdgcn_wmma_f32_16x16x32_bf16(false, a, false, b,
                                                  (short)0, acc, false, false);
  }
#pragma unroll
  for (int j = 0; j < 8; ++j) {
    int m = m0 + j + (hi ? 8 : 0);
    if (m < M) C[(long)m * DIM2 + n0 + l15] = fmaxf(acc[j], 0.f);
  }
}

// ---------------------------------------------------------------------------
// Highway fused: t = sigmoid(l1 @ Kw + bg);  out = t * relu(gpre) + (1-t) * l1
// M multiple of 16, inner dim fixed 256. Kw 16-col slice staged to LDS via
// async b128 loads; one wave per block.
// ---------------------------------------------------------------------------
__global__ void __launch_bounds__(32)
wmma_highway(const float* __restrict__ l1, const float* __restrict__ Kw,
             const float* __restrict__ bg, const float* __restrict__ gpre,
             float* __restrict__ out) {
  __shared__ float Bs[DIMD * 16];                // [k][n] slice, 16KB
  const int lane = threadIdx.x;
  const bool hi = lane >= 16;
  const int l15 = lane & 15;
  const int m0 = blockIdx.x * 16;
  const int n0 = blockIdx.y * 16;
  // stage Kw[:, n0:n0+16] -> Bs  (1024 float4s, 32 per lane)
  for (int i = lane; i < DIMD * 4; i += 32) {
    int row = i >> 2, part = (i & 3) * 4;
    stage_b128(Kw + (long)row * DIMD + n0 + part, Bs + row * 16 + part);
  }
  stage_wait();
  const float* Arow = l1 + (long)(m0 + l15) * DIMD;
  v8f acc = {};
#pragma unroll
  for (int k0 = 0; k0 < DIMD; k0 += 32) {
    const int kaLo = k0 + (hi ? 16 : 0);
    const int kaHi = k0 + (hi ? 24 : 8);
    f32x4 a0 = *(const f32x4*)(Arow + kaLo);
    f32x4 a1 = *(const f32x4*)(Arow + kaLo + 4);
    f32x4 a2 = *(const f32x4*)(Arow + kaHi);
    f32x4 a3 = *(const f32x4*)(Arow + kaHi + 4);
    v16bf a, b;
#pragma unroll
    for (int j = 0; j < 4; ++j) {
      a[j]      = (__bf16)a0[j];
      a[j + 4]  = (__bf16)a1[j];
      a[j + 8]  = (__bf16)a2[j];
      a[j + 12] = (__bf16)a3[j];
    }
    const int kb0 = k0 + (hi ? 16 : 0);
#pragma unroll
    for (int j = 0; j < 16; ++j)
      b[j] = (__bf16)Bs[(kb0 + j) * 16 + l15];
    acc = __builtin_amdgcn_wmma_f32_16x16x32_bf16(false, a, false, b,
                                                  (short)0, acc, false, false);
  }
#pragma unroll
  for (int j = 0; j < 8; ++j) {
    int m = m0 + j + (hi ? 8 : 0);
    int n = n0 + l15;
    float z = acc[j] + bg[n];
    float t = 1.f / (1.f + __expf(-z));
    long idx = (long)m * DIMD + n;
    float g = fmaxf(gpre[idx], 0.f);
    out[idx] = t * g + (1.f - t) * l1[idx];
  }
}

// ---------------------------------------------------------------------------
// rank-1 collapse of dense-attention scores:  v2[c] = sum_h w1[h,c]*w2[h]
// ---------------------------------------------------------------------------
__global__ void rank1_vec(const float* __restrict__ w1, const float* __restrict__ w2,
                          const float* __restrict__ w3, float* __restrict__ v2,
                          float* __restrict__ v3) {
  int c = blockIdx.x * blockDim.x + threadIdx.x;
  if (c >= DIM2) return;
  float s2 = 0.f, s3 = 0.f;
  for (int h = 0; h < HIDN; ++h) {
    float w = w1[(long)h * DIM2 + c];
    s2 += w * w2[h];
    s3 += w * w3[h];
  }
  v2[c] = s2;
  v3[c] = s3;
}

// f1[r] = feats[r,:]·v2 + b2 ; f2[r] = feats[r,:]·v3 + b3   (one wave per row)
__global__ void __launch_bounds__(32)
rowdot2(const float* __restrict__ feats, const float* __restrict__ v2,
        const float* __restrict__ v3, const float* __restrict__ b2,
        const float* __restrict__ b3, float* __restrict__ f1,
        float* __restrict__ f2) {
  int r = blockIdx.x;
  int lane = threadIdx.x;
  float s2 = 0.f, s3 = 0.f;
  for (int c = lane; c < DIM2; c += 32) {
    float f = feats[(long)r * DIM2 + c];
    s2 += f * v2[c];
    s3 += f * v3[c];
  }
  s2 = waveSum(s2);
  s3 = waveSum(s3);
  if (lane == 0) {
    f1[r] = s2 + b2[0];
    f2[r] = s3 + b3[0];
  }
}

// per-relation sparse score: ls[r] = leaky(dual[r,:]·w + b)
__global__ void __launch_bounds__(32)
relscore(const float* __restrict__ dual, const float* __restrict__ w,
         const float* __restrict__ b, float* __restrict__ ls) {
  int r = blockIdx.x;
  int lane = threadIdx.x;
  float s = 0.f;
  for (int c = lane; c < DIM2; c += 32) s += dual[(long)r * DIM2 + c] * w[c];
  s = waveSum(s);
  if (lane == 0) ls[r] = leakyf(s + b[0]);
}

// ---------------------------------------------------------------------------
// dense attention row softmax -> coefs [500, 512] (cols 500..511 zero pad)
// ---------------------------------------------------------------------------
__global__ void att_softmax(const float* __restrict__ adj, const float* __restrict__ f1,
                            const float* __restrict__ f2, float* __restrict__ coefs) {
  __shared__ float red[16];
  int r = blockIdx.x;
  int c = threadIdx.x;
  bool valid = c < R_REL;
  float a = valid ? adj[(long)r * R_REL + c] : 0.f;
  float lg = -1e30f;
  if (valid) {
    lg = leakyf(a * (f1[r] + f2[c]));
    if (!(a > 0.f)) lg += -1e9f;
  }
  float v = lg;
#pragma unroll
  for (int off = 16; off > 0; off >>= 1) v = fmaxf(v, __shfl_down(v, off, 32));
  if ((c & 31) == 0) red[c >> 5] = v;
  __syncthreads();
  if (c < 16) {
    float w = red[c];
#pragma unroll
    for (int off = 8; off > 0; off >>= 1) w = fmaxf(w, __shfl_down(w, off, 16));
    if (c == 0) red[0] = w;
  }
  __syncthreads();
  float mx = red[0];
  __syncthreads();
  float ex = valid ? __expf(lg - mx) : 0.f;
  float sv = ex;
#pragma unroll
  for (int off = 16; off > 0; off >>= 1) sv += __shfl_down(sv, off, 32);
  if ((c & 31) == 0) red[c >> 5] = sv;
  __syncthreads();
  if (c < 16) {
    float w = red[c];
#pragma unroll
    for (int off = 8; off > 0; off >>= 1) w += __shfl_down(w, off, 16);
    if (c == 0) red[0] = w;
  }
  __syncthreads();
  float s = red[0];
  coefs[(long)r * DIM2 + c] = valid ? (ex / s) : 0.f;
}

// ---------------------------------------------------------------------------
// sparse attention: segment max / exp-sum / scatter
// ---------------------------------------------------------------------------
__global__ void seg_max_k(const int* __restrict__ row, const int* __restrict__ rel,
                          const float* __restrict__ ls, float* __restrict__ m) {
  int e = blockIdx.x * blockDim.x + threadIdx.x;
  if (e >= N_EDGE) return;
  atomicMaxF(&m[row[e]], ls[rel[e]]);
}

__global__ void seg_expsum_k(const int* __restrict__ row, const int* __restrict__ rel,
                             const float* __restrict__ ls, const float* __restrict__ m,
                             float* __restrict__ ex, float* __restrict__ s) {
  int e = blockIdx.x * blockDim.x + threadIdx.x;
  if (e >= N_EDGE) return;
  int r = row[e];
  float v = __expf(ls[rel[e]] - m[r]);
  ex[e] = v;
  atomicAdd(&s[r], v);
}

// one block (64 threads) per edge; each thread handles 4 dims via b128 gather
__global__ void __launch_bounds__(64)
scatter_att_k(const int* __restrict__ row, const int* __restrict__ col,
              const float* __restrict__ ex, const float* __restrict__ s,
              const float* __restrict__ X, float* __restrict__ acc) {
  int e = blockIdx.x;
  int t = threadIdx.x << 2;
  int r = row[e], c = col[e];
  float coef = ex[e] / s[r];
  f32x4 v = *(const f32x4*)(X + (long)c * DIMD + t);
  float* dst = acc + (long)r * DIMD + t;
  atomicAdd(dst + 0, coef * v[0]);
  atomicAdd(dst + 1, coef * v[1]);
  atomicAdd(dst + 2, coef * v[2]);
  atomicAdd(dst + 3, coef * v[3]);
}

__global__ void __launch_bounds__(64)
scatter_gcn_k(const int* __restrict__ mr, const int* __restrict__ mc,
              const float* __restrict__ mv, const float* __restrict__ X,
              const float* __restrict__ Wg, float* __restrict__ acc) {
  int e = blockIdx.x;
  int t = threadIdx.x << 2;
  int r = mr[e], c = mc[e];
  float w = mv[e];
  f32x4 v = *(const f32x4*)(X + (long)c * DIMD + t);
  f32x4 g = *(const f32x4*)(Wg + t);
  float* dst = acc + (long)r * DIMD + t;
  atomicAdd(dst + 0, w * v[0] * g[0]);
  atomicAdd(dst + 1, w * v[1] * g[1]);
  atomicAdd(dst + 2, w * v[2] * g[2]);
  atomicAdd(dst + 3, w * v[3] * g[3]);
}

// out = X0 + alpha * relu(acc), vectorized
__global__ void axpy_relu4(const float* __restrict__ X0, const float* __restrict__ acc,
                           float alpha, float* __restrict__ out, long n4) {
  long i = (long)blockIdx.x * blockDim.x + threadIdx.x;
  if (i >= n4) return;
  f32x4 x = ((const f32x4*)X0)[i];
  f32x4 a = ((const f32x4*)acc)[i];
  f32x4 o;
#pragma unroll
  for (int j = 0; j < 4; ++j) o[j] = x[j] + alpha * fmaxf(a[j], 0.f);
  ((f32x4*)out)[i] = o;
}

// ---------------------------------------------------------------------------
extern "C" void kernel_launch(void* const* d_in, const int* in_sizes, int n_in,
                              void* d_out, int out_size, void* d_ws, size_t ws_size,
                              hipStream_t stream) {
  (void)in_sizes; (void)n_in; (void)out_size; (void)ws_size;
  const float* X0     = (const float*)d_in[0];
  const float* head_r = (const float*)d_in[1];
  const float* tail_r = (const float*)d_in[2];
  const float* dual_A = (const float*)d_in[3];
  const int*   e_row  = (const int*)d_in[4];
  const int*   e_col  = (const int*)d_in[5];
  const int*   e_rel  = (const int*)d_in[6];
  const int*   g_row  = (const int*)d_in[7];
  const int*   g_col  = (const int*)d_in[8];
  const float* g_val  = (const float*)d_in[9];
  const float* w_self1 = (const float*)d_in[10];
  const float* w_self2 = (const float*)d_in[11];
  const float* b_self2 = (const float*)d_in[12];
  const float* w_self3 = (const float*)d_in[13];
  const float* b_self3 = (const float*)d_in[14];
  const float* w_sp1   = (const float*)d_in[15];
  const float* b_sp1   = (const float*)d_in[16];
  const float* w_dual1 = (const float*)d_in[17];
  const float* w_dual2 = (const float*)d_in[18];
  const float* b_dual2 = (const float*)d_in[19];
  const float* w_dual3 = (const float*)d_in[20];
  const float* b_dual3 = (const float*)d_in[21];
  const float* w_sp2   = (const float*)d_in[22];
  const float* b_sp2   = (const float*)d_in[23];
  const float* w_gcn1  = (const float*)d_in[24];
  const float* w_gcn2  = (const float*)d_in[25];
  const float* k_hw1   = (const float*)d_in[26];
  const float* bg_hw1  = (const float*)d_in[27];
  const float* k_hw2   = (const float*)d_in[28];
  const float* bg_hw2  = (const float*)d_in[29];

  // workspace carve-up (floats); R-matrices padded to 512 rows x 512 cols
  float* ws = (float*)d_ws;
  size_t off = 0;
  auto alloc = [&](size_t n) { float* p = ws + off; off += n; return p; };
  float* hs    = alloc(512);
  float* ts    = alloc(512);
  float* dX    = alloc(512 * 512);              // dX1 then dX2 (zero-padded)
  float* v2    = alloc(512);
  float* v3    = alloc(512);
  float* f1    = alloc(512);
  float* f2    = alloc(512);
  float* coefs = alloc(512 * 512);              // col pad written by att_softmax
  float* dH1   = alloc(512 * 512);              // row pad zeroed once
  float* dH2   = alloc(512 * 512);
  float* ls    = alloc(512);
  float* segm  = alloc(E_ENT);
  float* segs  = alloc(E_ENT);
  float* exv   = alloc(N_EDGE);
  float* acc   = alloc((size_t)E_ENT * DIMD);   // scatter accumulator (reused 4x)
  float* bufA  = alloc((size_t)E_ENT * DIMD);   // pX1, later h1
  float* bufB  = alloc((size_t)E_ENT * DIMD);   // pX2

  const long NXD = (long)E_ENT * DIMD;
  const int  KPB = 6272;                        // 196*32, x8 z-blocks covers 50000
  const dim3 tileATB((R_REL + 15) / 16, DIMD / 32, 8);  // 32 x 8 x 8
  const dim3 tileRR((R_REL + 15) / 16, DIM2 / 16);      // 32 x 32
  const dim3 tileHW(E_ENT / 16, DIMD / 16);             // 3125 x 16
  const int NEB = (N_EDGE + 255) / 256;
  const int EEB = (E_ENT + 255) / 256;
  const int XDB = (int)((NXD + 255) / 256);
  const int XDB4 = (int)((NXD / 4 + 255) / 256);

  // --- one-time zeroing: column sums + dH1 row pad ---
  fill_f32<<<4, 256, 0, stream>>>(hs, 1024, 0.f);       // hs+ts contiguous
  fill_f32<<<(12 * 512 + 255) / 256, 256, 0, stream>>>(dH1 + 500 * DIM2, 12 * DIM2, 0.f);
  colsum2<<<dim3(2, 64), 256, 0, stream>>>(head_r, tail_r, hs, ts);

  // --- layer 1: dX1 = compute_r(X0) ---
  fill_f32<<<(512 * 512 + 255) / 256, 256, 0, stream>>>(dX, 512 * 512, 0.f);
  wmma_atb_atomic<<<tileATB, 32, 0, stream>>>(head_r, X0, dX, 0, KPB);
  wmma_atb_atomic<<<tileATB, 32, 0, stream>>>(tail_r, X0, dX, DIMD, KPB);
  scale_dx<<<R_REL, 512, 0, stream>>>(dX, hs, ts);

  // dense attention #1: dH1 = dense_att(dX1, dX1)
  rank1_vec<<<2, 256, 0, stream>>>(w_self1, w_self2, w_self3, v2, v3);
  rowdot2<<<R_REL, 32, 0, stream>>>(dX, v2, v3, b_self2, b_self3, f1, f2);
  att_softmax<<<R_REL, 512, 0, stream>>>(dual_A, f1, f2, coefs);
  wmma_gemm_relu<<<tileRR, 32, 0, stream>>>(coefs, dX, dH1, R_REL);

  // sparse attention #1: pX1 = X0 + ALPHA * relu(scatter)
  relscore<<<R_REL, 32, 0, stream>>>(dH1, w_sp1, b_sp1, ls);
  fill_f32<<<EEB, 256, 0, stream>>>(segm, E_ENT, -INFINITY);
  fill_f32<<<EEB, 256, 0, stream>>>(segs, E_ENT, 0.f);
  fill_f32<<<XDB, 256, 0, stream>>>(acc, NXD, 0.f);
  seg_max_k<<<NEB, 256, 0, stream>>>(e_row, e_rel, ls, segm);
  seg_expsum_k<<<NEB, 256, 0, stream>>>(e_row, e_rel, ls, segm, exv, segs);
  scatter_att_k<<<N_EDGE, 64, 0, stream>>>(e_row, e_col, exv, segs, X0, acc);
  axpy_relu4<<<XDB4, 256, 0, stream>>>(X0, acc, ALPHA_C, bufA, NXD / 4);

  // --- layer 2: dX2 = compute_r(pX1) ---
  fill_f32<<<(512 * 512 + 255) / 256, 256, 0, stream>>>(dX, 512 * 512, 0.f);
  wmma_atb_atomic<<<tileATB, 32, 0, stream>>>(head_r, bufA, dX, 0, KPB);
  wmma_atb_atomic<<<tileATB, 32, 0, stream>>>(tail_r, bufA, dX, DIMD, KPB);
  scale_dx<<<R_REL, 512, 0, stream>>>(dX, hs, ts);

  // dense attention #2: dH2 = dense_att(dH1, dX2)
  rank1_vec<<<2, 256, 0, stream>>>(w_dual1, w_dual2, w_dual3, v2, v3);
  rowdot2<<<R_REL, 32, 0, stream>>>(dX, v2, v3, b_dual2, b_dual3, f1, f2);
  att_softmax<<<R_REL, 512, 0, stream>>>(dual_A, f1, f2, coefs);
  wmma_gemm_relu<<<tileRR, 32, 0, stream>>>(coefs, dH1, dH2, R_REL);

  // sparse attention #2: pX2 = X0 + BETA * relu(scatter over pX1)
  relscore<<<R_REL, 32, 0, stream>>>(dH2, w_sp2, b_sp2, ls);
  fill_f32<<<EEB, 256, 0, stream>>>(segm, E_ENT, -INFINITY);
  fill_f32<<<EEB, 256, 0, stream>>>(segs, E_ENT, 0.f);
  fill_f32<<<XDB, 256, 0, stream>>>(acc, NXD, 0.f);
  seg_max_k<<<NEB, 256, 0, stream>>>(e_row, e_rel, ls, segm);
  seg_expsum_k<<<NEB, 256, 0, stream>>>(e_row, e_rel, ls, segm, exv, segs);
  scatter_att_k<<<N_EDGE, 64, 0, stream>>>(e_row, e_col, exv, segs, bufA, acc);
  axpy_relu4<<<XDB4, 256, 0, stream>>>(X0, acc, BETA_C, bufB, NXD / 4);

  // --- GCN1 + highway1:  h1 (bufA) = highway(pX2, relu(gcn(pX2))) ---
  fill_f32<<<XDB, 256, 0, stream>>>(acc, NXD, 0.f);
  scatter_gcn_k<<<N_EDGE, 64, 0, stream>>>(g_row, g_col, g_val, bufB, w_gcn1, acc);
  wmma_highway<<<tileHW, 32, 0, stream>>>(bufB, k_hw1, bg_hw1, acc, bufA);

  // --- GCN2 + highway2:  out = highway(h1, relu(gcn(h1))) ---
  fill_f32<<<XDB, 256, 0, stream>>>(acc, NXD, 0.f);
  scatter_gcn_k<<<N_EDGE, 64, 0, stream>>>(g_row, g_col, g_val, bufA, w_gcn2, acc);
  wmma_highway<<<tileHW, 32, 0, stream>>>(bufA, k_hw2, bg_hw2, acc, (float*)d_out);
}
